// RNN_85341000172340
// MI455X (gfx1250) — matrix operand order repaired and checked by reading, not compiled, not run on hardware
//
#include <hip/hip_runtime.h>
#include <hip/hip_bf16.h>

typedef __attribute__((ext_vector_type(16))) _Float16 v16h;
typedef __attribute__((ext_vector_type(8)))  _Float16 v8h;
typedef __attribute__((ext_vector_type(4)))  _Float16 v4h;
typedef __attribute__((ext_vector_type(8)))  float    v8f;
typedef __attribute__((ext_vector_type(4)))  float    v4f;

#define B_ 512
#define T_ 512
#define I_ 128
#define H_ 512
#define C_ 128
#define OUTT_ 128
#define LDA 648   // padded f16 LDS row stride (uniform 2-way bank pattern for A-frag reads)
#define LDM 520   // padded f16 LDS row stride for mid tile

__device__ __forceinline__ v8f wmma_f16(v16h a, v16h b, v8f c) {
  // D = A(16x32 f16) x B(32x16 f16) + C(16x16 f32)
  return __builtin_amdgcn_wmma_f32_16x16x32_f16(
      /*neg_a=*/false, a, /*neg_b=*/false, b,
      /*c_mod=*/(short)0, c, /*reuse_a=*/false, /*reuse_b=*/false);
}

// Branch-free tanh on the TRANS pipe (co-executes with WMMA/VALU on CDNA5).
// Prefer hardware v_tanh_f32 when the toolchain exposes it; otherwise
// tanh(x) = 1 - 2/(exp(2x)+1) via v_exp_f32 + v_rcp_f32 (saturates correctly at +/-inf).
__device__ __forceinline__ float fast_tanhf(float x) {
#if __has_builtin(__builtin_amdgcn_tanhf)
  return __builtin_amdgcn_tanhf(x);
#else
  float e = __builtin_amdgcn_exp2f(x * 2.885390081777926814f);  // exp(2x) = 2^(2x*log2 e)
  return 1.0f - 2.0f * __builtin_amdgcn_rcpf(e + 1.0f);
#endif
}

// A-fragment per documented 16-bit A layout:
// lane L: row m = L&15; elements 0..7 -> K = half*8 + e, elements 8..15 -> K = 16 + half*8 + e
__device__ __forceinline__ v16h load_a_frag(const _Float16* As, int lda, int lane, int k0) {
  int m    = lane & 15;
  int half = lane >> 4;
  const _Float16* row = As + m * lda;
  v8h lo = *(const v8h*)(row + k0 + half * 8);
  v8h hi = *(const v8h*)(row + k0 + 16 + half * 8);
  v16h a;
#pragma unroll
  for (int i = 0; i < 8; ++i) { a[i] = lo[i]; a[i + 8] = hi[i]; }
  return a;
}

// Pack weight matrix rows into WMMA B-fragment order.
// frag_id = kc*NT + nt ; element: lane L -> n = nt*16 + (L&15), k = kc*32 + (e<8 ? (L>>4)*8 + e : 16 + (L>>4)*8 + e-8)
// Value: k <  Ksplit -> W0[n*ld0 + k]        (W_hh rows)
//        k >= Ksplit -> W1[n*ld1 + k-Ksplit] (W_ih rows, encoder concat)
__global__ void pack_frag_kernel(const float* __restrict__ W0, const float* __restrict__ W1,
                                 _Float16* __restrict__ dst, int NT, int KC, int Ksplit,
                                 int ld0, int ld1) {
  int idx = blockIdx.x * 256 + threadIdx.x;
  int total = NT * KC * 32 * 16;
  if (idx >= total) return;
  int e    = idx & 15;
  int lane = (idx >> 4) & 31;
  int frag = idx >> 9;          // kc*NT + nt
  int nt = frag % NT;
  int kc = frag / NT;
  int k = kc * 32 + ((e < 8) ? ((lane >> 4) * 8 + e) : (16 + (lane >> 4) * 8 + (e - 8)));
  int n = nt * 16 + (lane & 15);
  float v = (k < Ksplit) ? W0[(size_t)n * ld0 + k] : W1[(size_t)n * ld1 + (k - Ksplit)];
  dst[idx] = (_Float16)v;
}

__global__ void init_kernel(const float* __restrict__ b_ih, const float* __restrict__ b_hh,
                            float* __restrict__ bias2, _Float16* __restrict__ h0) {
  int i = blockIdx.x * 256 + threadIdx.x;
  if (i < H_) bias2[i] = b_ih[i] + b_hh[i];
  if (i < B_ * H_) h0[i] = (_Float16)0.f;
}

// One encoder time step: h_out = m ? tanh([h|x_t]@[W_hh|W_ih]^T + b) : h_in
// grid: 32 m_tiles x 4 n_quarters = 128 blocks, 4 waves/block, 2 n-tiles (16x32 C) per wave.
__global__ __launch_bounds__(128) void enc_step_kernel(
    const _Float16* __restrict__ hin, _Float16* __restrict__ hout,
    const _Float16* __restrict__ Wp, const float* __restrict__ x,
    const float* __restrict__ bias2, const int* __restrict__ lengths, int t) {
  __shared__ alignas(16) _Float16 As[16 * LDA];
  int tid = threadIdx.x, lane = tid & 31, wave = tid >> 5;
  int m_tile = blockIdx.x >> 2;
  int nq     = blockIdx.x & 3;
  int m0 = m_tile * 16;

  // Stage h rows (16 x 512 f16), coalesced 16B chunks
#pragma unroll
  for (int it = 0; it < 8; ++it) {
    int idx = tid + it * 128;             // 1024 v8h chunks
    int r = idx >> 6, c8 = idx & 63;
    *(v8h*)(As + r * LDA + c8 * 8) = *(const v8h*)(hin + (size_t)(m0 + r) * H_ + c8 * 8);
  }
  // Stage x_t rows (16 x 128 f32 -> f16) into cols 512..639
#pragma unroll
  for (int it = 0; it < 4; ++it) {
    int idx = tid + it * 128;             // 512 float4 chunks
    int r = idx >> 5, c4 = idx & 31;
    v4f v = *(const v4f*)(x + (size_t)(m0 + r) * T_ * I_ + (size_t)t * I_ + c4 * 4);
    v4h h4;
#pragma unroll
    for (int i = 0; i < 4; ++i) h4[i] = (_Float16)v[i];
    *(v4h*)(As + r * LDA + 512 + c4 * 4) = h4;
  }
  __syncthreads();

  int nt0 = nq * 8 + wave * 2;            // first of 2 n-tiles for this wave
  v8f c0 = {}, c1 = {};
#pragma unroll 4
  for (int kc = 0; kc < 20; ++kc) {       // K = 640 = 512 (h) + 128 (x)
    v16h a = load_a_frag(As, LDA, lane, kc * 32);
    const _Float16* bbase = Wp + (size_t)(kc * 32 + nt0) * 512 + lane * 16;
    c0 = wmma_f16(a, *(const v16h*)(bbase), c0);
    c1 = wmma_f16(a, *(const v16h*)(bbase + 512), c1);
  }

  int nl = lane & 15, halfm = lane >> 4;
  // This lane's 8 output rows are m = halfm*8 .. halfm*8+7 -> two vectorized length loads
  const int4* lp = (const int4*)(lengths + m0 + halfm * 8);
  int4 len0 = lp[0], len1 = lp[1];
  int lens[8] = {len0.x, len0.y, len0.z, len0.w, len1.x, len1.y, len1.z, len1.w};

  v8f accs[2] = {c0, c1};
#pragma unroll
  for (int j = 0; j < 2; ++j) {
    int n = (nt0 + j) * 16 + nl;
    float bias = bias2[n];
#pragma unroll
    for (int r = 0; r < 8; ++r) {
      int m = halfm * 8 + r;              // C layout: lanes 16-31 hold rows m=8..15
      int mg = m0 + m;
      float nh  = fast_tanhf(accs[j][r] + bias);
      float old = (float)As[m * LDA + n];
      float val = (t < lens[r]) ? nh : old;
      hout[(size_t)mg * H_ + n] = (_Float16)val;
    }
  }
}

// One decoder recurrence step: nh = tanh(h@W_hh^T + b); h_out = nh; also record nh for post pass.
__global__ __launch_bounds__(128) void dec_step_kernel(
    const _Float16* __restrict__ hin, _Float16* __restrict__ hout,
    _Float16* __restrict__ nh_slice, const _Float16* __restrict__ Wp,
    const float* __restrict__ bias2) {
  __shared__ alignas(16) _Float16 As[16 * LDA];
  int tid = threadIdx.x, lane = tid & 31, wave = tid >> 5;
  int m_tile = blockIdx.x >> 2;
  int nq     = blockIdx.x & 3;
  int m0 = m_tile * 16;

#pragma unroll
  for (int it = 0; it < 8; ++it) {
    int idx = tid + it * 128;
    int r = idx >> 6, c8 = idx & 63;
    *(v8h*)(As + r * LDA + c8 * 8) = *(const v8h*)(hin + (size_t)(m0 + r) * H_ + c8 * 8);
  }
  __syncthreads();

  int nt0 = nq * 8 + wave * 2;
  v8f c0 = {}, c1 = {};
#pragma unroll 4
  for (int kc = 0; kc < 16; ++kc) {       // K = 512, reuses first 16 k-chunks of packed Wenc
    v16h a = load_a_frag(As, LDA, lane, kc * 32);
    const _Float16* bbase = Wp + (size_t)(kc * 32 + nt0) * 512 + lane * 16;
    c0 = wmma_f16(a, *(const v16h*)(bbase), c0);
    c1 = wmma_f16(a, *(const v16h*)(bbase + 512), c1);
  }

  int nl = lane & 15, halfm = lane >> 4;
  v8f accs[2] = {c0, c1};
#pragma unroll
  for (int j = 0; j < 2; ++j) {
    int n = (nt0 + j) * 16 + nl;
    float bias = bias2[n];
#pragma unroll
    for (int r = 0; r < 8; ++r) {
      int m = halfm * 8 + r;
      int mg = m0 + m;
      _Float16 nh = (_Float16)fast_tanhf(accs[j][r] + bias);
      hout[(size_t)mg * H_ + n]     = nh;
      nh_slice[(size_t)mg * H_ + n] = nh;
    }
  }
}

// Batched decoder head over all (t,b) rows: mid = relu(nh@fc1^T + b1); out = mid@fc2^T + b2.
// grid: 65536/16 = 4096 blocks, 4 waves; phase1: wave does 16x128 of mid, phase2: 16x32 of out.
__global__ __launch_bounds__(128) void dec_post_kernel(
    const _Float16* __restrict__ nh_all, const _Float16* __restrict__ fc1p,
    const _Float16* __restrict__ fc2p, const float* __restrict__ fc1b,
    const float* __restrict__ fc2b, float* __restrict__ out) {
  __shared__ alignas(16) _Float16 As[16 * LDA];
  __shared__ alignas(16) _Float16 Ms[16 * LDM];
  int tid = threadIdx.x, lane = tid & 31, wave = tid >> 5;
  size_t r0 = (size_t)blockIdx.x * 16;    // row = t*B + b in nh_all
  int nl = lane & 15, halfm = lane >> 4;

#pragma unroll
  for (int it = 0; it < 8; ++it) {
    int idx = tid + it * 128;
    int r = idx >> 6, c8 = idx & 63;
    *(v8h*)(As + r * LDA + c8 * 8) = *(const v8h*)(nh_all + (r0 + r) * H_ + c8 * 8);
  }
  __syncthreads();

  // Phase 1: mid(16x512) = relu(nh @ fc1^T + fc1_b) -> LDS (f16)
  {
    v8f acc[8] = {};
    for (int kc = 0; kc < 16; ++kc) {
      v16h a = load_a_frag(As, LDA, lane, kc * 32);
      const _Float16* bbase = fc1p + (size_t)(kc * 32 + wave * 8) * 512 + lane * 16;
#pragma unroll
      for (int j = 0; j < 8; ++j)
        acc[j] = wmma_f16(a, *(const v16h*)(bbase + (size_t)j * 512), acc[j]);
    }
#pragma unroll
    for (int j = 0; j < 8; ++j) {
      int n = (wave * 8 + j) * 16 + nl;
      float bias = fc1b[n];
#pragma unroll
      for (int r = 0; r < 8; ++r) {
        int m = halfm * 8 + r;
        float v = acc[j][r] + bias;
        Ms[m * LDM + n] = (_Float16)fmaxf(v, 0.f);
      }
    }
  }
  __syncthreads();

  // Phase 2: out(16x128) = mid @ fc2^T + fc2_b, write f32 in (b, t, c) order
  {
    v8f a0 = {}, a1 = {};
    for (int kc = 0; kc < 16; ++kc) {
      v16h a = load_a_frag(Ms, LDM, lane, kc * 32);
      const _Float16* bbase = fc2p + (size_t)(kc * 8 + wave * 2) * 512 + lane * 16;
      a0 = wmma_f16(a, *(const v16h*)(bbase), a0);
      a1 = wmma_f16(a, *(const v16h*)(bbase + 512), a1);
    }
    v8f accs[2] = {a0, a1};
#pragma unroll
    for (int j = 0; j < 2; ++j) {
      int c = (wave * 2 + j) * 16 + nl;
      float bias = fc2b[c];
#pragma unroll
      for (int r = 0; r < 8; ++r) {
        int m = halfm * 8 + r;
        size_t rg = r0 + m;                 // rg = t*B + b
        int tt = (int)(rg >> 9);            // / 512
        int bb = (int)(rg & 511);
        out[((size_t)bb * OUTT_ + tt) * C_ + c] = accs[j][r] + bias;
      }
    }
  }
}

extern "C" void kernel_launch(void* const* d_in, const int* in_sizes, int n_in,
                              void* d_out, int out_size, void* d_ws, size_t ws_size,
                              hipStream_t stream) {
  (void)in_sizes; (void)n_in; (void)out_size; (void)ws_size;
  const float* x       = (const float*)d_in[0];
  const int*   lengths = (const int*)d_in[1];
  /* d_in[2] = out_lengths scalar == 128 (compile-time constant OUTT_) */
  const float* W_ih    = (const float*)d_in[3];
  const float* W_hh    = (const float*)d_in[4];
  const float* b_ih    = (const float*)d_in[5];
  const float* b_hh    = (const float*)d_in[6];
  const float* fc1_W   = (const float*)d_in[7];
  const float* fc1_b   = (const float*)d_in[8];
  const float* fc2_W   = (const float*)d_in[9];
  const float* fc2_b   = (const float*)d_in[10];
  float* out = (float*)d_out;

  char* ws = (char*)d_ws;
  size_t off = 0;
  auto alloc = [&](size_t bytes) -> void* {
    void* p = ws + off;
    off = (off + bytes + 255) & ~(size_t)255;
    return p;
  };
  _Float16* Wenc   = (_Float16*)alloc((size_t)640 * 512 * 2);   // [W_hh|W_ih] fragment-packed
  _Float16* fc1p   = (_Float16*)alloc((size_t)512 * 512 * 2);
  _Float16* fc2p   = (_Float16*)alloc((size_t)512 * 128 * 2);
  float*    bias2  = (float*)   alloc((size_t)H_ * 4);
  _Float16* hA     = (_Float16*)alloc((size_t)B_ * H_ * 2);
  _Float16* hB     = (_Float16*)alloc((size_t)B_ * H_ * 2);
  _Float16* nh_all = (_Float16*)alloc((size_t)OUTT_ * B_ * H_ * 2);

  pack_frag_kernel<<<(640 * 512 + 255) / 256, 256, 0, stream>>>(W_hh, W_ih, Wenc, 32, 20, 512, 512, 128);
  pack_frag_kernel<<<(512 * 512 + 255) / 256, 256, 0, stream>>>(fc1_W, fc1_W, fc1p, 32, 16, 512, 512, 512);
  pack_frag_kernel<<<(512 * 128 + 255) / 256, 256, 0, stream>>>(fc2_W, fc2_W, fc2p, 8, 16, 512, 512, 512);
  init_kernel<<<(B_ * H_ + 255) / 256, 256, 0, stream>>>(b_ih, b_hh, bias2, hA);

  _Float16* cur = hA;
  _Float16* nxt = hB;
  for (int t = 0; t < T_; ++t) {
    enc_step_kernel<<<128, 128, 0, stream>>>(cur, nxt, Wenc, x, bias2, lengths, t);
    _Float16* tmp = cur; cur = nxt; nxt = tmp;
  }
  for (int t = 0; t < OUTT_; ++t) {
    dec_step_kernel<<<128, 128, 0, stream>>>(cur, nxt, nh_all + (size_t)t * B_ * H_, Wenc, bias2);
    _Float16* tmp = cur; cur = nxt; nxt = tmp;
  }
  dec_post_kernel<<<(OUTT_ * B_) / 16, 128, 0, stream>>>(nh_all, fc1p, fc2p, fc1_b, fc2_b, out);
}
